// CrossAttentionBlock_9740985827775
// MI455X (gfx1250) — matrix operand order repaired and checked by reading, not compiled, not run on hardware
//
#include <hip/hip_runtime.h>

typedef _Float16 half_t;
typedef __attribute__((ext_vector_type(16))) _Float16 v16h;
typedef __attribute__((ext_vector_type(8)))  _Float16 v8h;
typedef __attribute__((ext_vector_type(8)))  float    v8f;
typedef __attribute__((ext_vector_type(4)))  float    v4f;

#define D_MODEL 256
#define N_HEADS 8
#define D_HEAD  32
#define NQ      1024
#define NKV     8192
#define BATCH   2
#define KV_SPLIT 4            /* kv-dimension split for occupancy */
#define ATTN_SCALE 0.17677669529663689f  /* 32^-0.5 */
#define LN_EPS  1e-5f

// ---------------------------------------------------------------------------
// Fragment helpers (wave32 WMMA 16x16x32 f16).
// A 16x32 f16: lane L holds row M=L%16; element i holds
//   K = i + 16*(L/16)  -> two adjacent 16B chunks per lane.
// B 32x16 f16: lane L holds col N=L%16; element i holds K = i + 16*(L/16)
//   -> one contiguous 32B chunk per lane.
// C/D f32: lane L holds col N=L%16; VGPR j holds row M = j + 8*(L/16).
// ---------------------------------------------------------------------------
__device__ __forceinline__ v16h load_frag_a(const half_t* base, int g) {
  v8h lo = *(const v8h*)(base + g * 16);
  v8h hi = *(const v8h*)(base + g * 16 + 8);
  v16h r;
#pragma unroll
  for (int i = 0; i < 8; ++i) { r[i] = lo[i]; r[i + 8] = hi[i]; }
  return r;
}

__device__ __forceinline__ v8f wmma_f16(v16h a, v16h b, v8f c) {
  return __builtin_amdgcn_wmma_f32_16x16x32_f16(false, a, false, b, (short)0, c,
                                                false, false);
}

// Half-wave swap (xor 16) without LDS: v_permlanex16_b32 with identity selects.
__device__ __forceinline__ float xor16f(float x) {
  int r = __builtin_amdgcn_permlanex16(__float_as_int(x), __float_as_int(x),
                                       0x76543210, 0xfedcba98, false, false);
  return __int_as_float(r);
}

// ---------------------------------------------------------------------------
// Weight convert f32 -> f16
// ---------------------------------------------------------------------------
__global__ void cvt_kernel(const float* __restrict__ x, half_t* __restrict__ y, int n) {
  int i = blockIdx.x * blockDim.x + threadIdx.x;
  if (i < n) y[i] = (half_t)x[i];
}

// ---------------------------------------------------------------------------
// LayerNorm: one wave per 256-elem row, f16 output.
// ---------------------------------------------------------------------------
__global__ void ln_kernel(const float* __restrict__ x, const float* __restrict__ gamma,
                          const float* __restrict__ beta, half_t* __restrict__ y,
                          int nrows) {
  const int lane = threadIdx.x & 31;
  const int row  = (blockIdx.x * blockDim.x + threadIdx.x) >> 5;
  if (row >= nrows) return;
  const float* xr = x + (size_t)row * D_MODEL + lane * 8;
  v4f a = *(const v4f*)(xr);
  v4f b = *(const v4f*)(xr + 4);
  float v[8];
#pragma unroll
  for (int i = 0; i < 4; ++i) { v[i] = a[i]; v[i + 4] = b[i]; }
  float s = 0.f;
#pragma unroll
  for (int i = 0; i < 8; ++i) s += v[i];
#pragma unroll
  for (int m = 1; m < 32; m <<= 1) s += __shfl_xor(s, m, 32);
  const float mean = s * (1.0f / D_MODEL);
  float vs = 0.f;
#pragma unroll
  for (int i = 0; i < 8; ++i) { float d = v[i] - mean; vs += d * d; }
#pragma unroll
  for (int m = 1; m < 32; m <<= 1) vs += __shfl_xor(vs, m, 32);
  const float rstd = rsqrtf(vs * (1.0f / D_MODEL) + LN_EPS);
  v4f gA = *(const v4f*)(gamma + lane * 8);
  v4f gB = *(const v4f*)(gamma + lane * 8 + 4);
  v4f bA = *(const v4f*)(beta + lane * 8);
  v4f bB = *(const v4f*)(beta + lane * 8 + 4);
  v8h o;
#pragma unroll
  for (int i = 0; i < 4; ++i) {
    o[i]     = (half_t)((v[i]     - mean) * rstd * gA[i] + bA[i]);
    o[i + 4] = (half_t)((v[i + 4] - mean) * rstd * gB[i] + bB[i]);
  }
  *(v8h*)(y + (size_t)row * D_MODEL + lane * 8) = o;
}

// ---------------------------------------------------------------------------
// GEMM: Y[m,n] = sum_k X[m,k]*W[n,k]  (nn.Linear, X f16, W f16).
// grid.x = M/16, block = 512 (16 waves); wave w owns n-tile [16w, 16w+16).
// MODE 0: store f16 row-major [M][256]            (Q, K projections)
// MODE 1: store f16 transposed [B][H][32][NKV]    (V projection)
// MODE 2: store f32 row-major + residual add      (output projection)
// ---------------------------------------------------------------------------
template <int MODE>
__global__ void gemm_kernel(const half_t* __restrict__ X, const half_t* __restrict__ W,
                            half_t* __restrict__ Y, const float* __restrict__ resid,
                            float* __restrict__ Yf) {
  const int lane = threadIdx.x & 31;
  const int wave = threadIdx.x >> 5;
  const int lr = lane & 15;
  const int g  = (lane >> 4) & 1;
  const int m0 = blockIdx.x * 16;
  const int n0 = wave * 16;

  const half_t* arow = X + (size_t)(m0 + lr) * D_MODEL;
  const half_t* brow = W + (size_t)(n0 + lr) * D_MODEL;

  v8f acc = {};
#pragma unroll
  for (int k0 = 0; k0 < D_MODEL; k0 += 32) {
    v16h a = load_frag_a(arow + k0, g);
    v16h b = *(const v16h*)(brow + k0 + 16 * g);
    acc = wmma_f16(a, b, acc);
  }

  if (MODE == 0) {
#pragma unroll
    for (int j = 0; j < 8; ++j)
      Y[(size_t)(m0 + j + 8 * g) * D_MODEL + n0 + lr] = (half_t)acc[j];
  } else if (MODE == 1) {
    const int n   = n0 + lr;           // n = h*32 + d
    const int bb  = m0 / NKV;          // batch
    const int kvb = (m0 % NKV) + 8 * g;
    v8h r;
#pragma unroll
    for (int j = 0; j < 8; ++j) r[j] = (half_t)acc[j];
    *(v8h*)(Y + ((size_t)(bb * N_HEADS + (n >> 5)) * D_HEAD + (n & 31)) * NKV + kvb) = r;
  } else {
#pragma unroll
    for (int j = 0; j < 8; ++j) {
      size_t idx = (size_t)(m0 + j + 8 * g) * D_MODEL + n0 + lr;
      Yf[idx] = resid[idx] + acc[j];
    }
  }
}

// ---------------------------------------------------------------------------
// Flash attention, one wave per (kv-split, b, h, 16-query tile).
// Sᵀ = K·Qᵀ (score tiles transposed), Oᵀ += Vᵀ·Pᵀ — avoids lane transposes.
// Writes unnormalized partial Oᵀ plus per-query (m, l) for the combine pass.
// ---------------------------------------------------------------------------
__global__ void attn_kernel(const half_t* __restrict__ Qp, const half_t* __restrict__ Kp,
                            const half_t* __restrict__ Vt, const float* __restrict__ bias,
                            float* __restrict__ Po, float* __restrict__ Pm,
                            float* __restrict__ Pl) {
  const int lane = threadIdx.x & 31;
  const int wave = threadIdx.x >> 5;
  const int lr = lane & 15;
  const int g  = (lane >> 4) & 1;
  const int sp = blockIdx.x >> 3;                  // kv split index
  const int q0 = ((blockIdx.x & 7) * 8 + wave) * 16;
  const int h  = blockIdx.y;
  const int b  = blockIdx.z;
  const int kv_lo = sp * (NKV / KV_SPLIT);
  const int kv_hi = kv_lo + (NKV / KV_SPLIT);

  // Q as B-fragment: lane holds query column q0+lr; elements over d = i + 16g.
  const v16h qf =
      *(const v16h*)(Qp + (size_t)(b * NQ + q0 + lr) * D_MODEL + h * D_HEAD + 16 * g);

  const half_t* Kb    = Kp + (size_t)b * NKV * D_MODEL + h * D_HEAD;
  const half_t* Vb    = Vt + (size_t)(b * N_HEADS + h) * D_HEAD * NKV;
  const float*  biasq = bias + ((size_t)(b * N_HEADS + h) * NQ + q0 + lr) * NKV;

  v8f o0 = {}, o1 = {};
  float mrun = -1e30f, lrun = 0.0f;

  for (int kv0 = kv_lo; kv0 < kv_hi; kv0 += 32) {
    __builtin_prefetch(biasq + kv0 + 256, 0, 0);  // ~8 chunks ahead on the stream

    // K as A-fragments (rows = kv), one WMMA per 16x16 transposed score tile.
    v16h kf0 = load_frag_a(Kb + (size_t)(kv0 + lr) * D_MODEL, g);
    v16h kf1 = load_frag_a(Kb + (size_t)(kv0 + 16 + lr) * D_MODEL, g);
    v8f c0 = {}, c1 = {};
    c0 = wmma_f16(kf0, qf, c0);  // Sᵀ rows kv0 + j + 8g
    c1 = wmma_f16(kf1, qf, c1);  // Sᵀ rows kv0 + 16 + j + 8g

    // bias[b][h][q][kv]: per lane 8 consecutive kv values per tile.
    v4f ba = *(const v4f*)(biasq + kv0 + 8 * g);
    v4f bb = *(const v4f*)(biasq + kv0 + 8 * g + 4);
    v4f bc = *(const v4f*)(biasq + kv0 + 16 + 8 * g);
    v4f bd = *(const v4f*)(biasq + kv0 + 16 + 8 * g + 4);

    float s0[8], s1[8];
#pragma unroll
    for (int j = 0; j < 4; ++j) {
      s0[j]     = fmaf(c0[j],     ATTN_SCALE, ba[j]);
      s0[j + 4] = fmaf(c0[j + 4], ATTN_SCALE, bb[j]);
      s1[j]     = fmaf(c1[j],     ATTN_SCALE, bc[j]);
      s1[j + 4] = fmaf(c1[j + 4], ATTN_SCALE, bd[j]);
    }

    // Online softmax over kv (per query column): 16 local values + partner half.
    float mt = s0[0];
#pragma unroll
    for (int j = 1; j < 8; ++j) mt = fmaxf(mt, s0[j]);
#pragma unroll
    for (int j = 0; j < 8; ++j) mt = fmaxf(mt, s1[j]);
    mt = fmaxf(mt, xor16f(mt));
    const float mnew = fmaxf(mrun, mt);
    const float corr = __expf(mrun - mnew);

    float p0[8], p1[8], lsum = 0.0f;
#pragma unroll
    for (int j = 0; j < 8; ++j) {
      p0[j] = __expf(s0[j] - mnew); lsum += p0[j];
      p1[j] = __expf(s1[j] - mnew); lsum += p1[j];
    }
    lsum += xor16f(lsum);
    lrun = lrun * corr + lsum;
    mrun = mnew;

#pragma unroll
    for (int j = 0; j < 8; ++j) { o0[j] *= corr; o1[j] *= corr; }

    // Build Pᵀ B-fragment: lane needs all 16 kv rows of its half's tile;
    // 8 rows are local, 8 live in the partner lane (permlanex16).
    float p0x[8], p1x[8];
#pragma unroll
    for (int j = 0; j < 8; ++j) {
      p0x[j] = xor16f(p0[j]);
      p1x[j] = xor16f(p1[j]);
    }
    v16h pf;
#pragma unroll
    for (int i = 0; i < 8; ++i) {
      pf[i]     = (half_t)(g ? p1x[i] : p0[i]);
      pf[i + 8] = (half_t)(g ? p1[i]  : p0x[i]);
    }

    // Vᵀ A-fragments: rows d (0..15 and 16..31), contiguous kv per lane.
    v16h vf0 = load_frag_a(Vb + (size_t)lr * NKV + kv0, g);
    v16h vf1 = load_frag_a(Vb + (size_t)(16 + lr) * NKV + kv0, g);
    o0 = wmma_f16(vf0, pf, o0);  // Oᵀ rows d = j + 8g
    o1 = wmma_f16(vf1, pf, o1);  // Oᵀ rows d = 16 + j + 8g
  }

  // Store unnormalized partials: Po[sp][b][h][q][d], (m,l) per query.
  const size_t OS = (size_t)BATCH * N_HEADS * NQ * D_HEAD;
  float* prow = Po + sp * OS + ((size_t)(b * N_HEADS + h) * NQ + q0 + lr) * D_HEAD;
  v4f w0a, w0b, w1a, w1b;
#pragma unroll
  for (int j = 0; j < 4; ++j) {
    w0a[j] = o0[j]; w0b[j] = o0[j + 4];
    w1a[j] = o1[j]; w1b[j] = o1[j + 4];
  }
  *(v4f*)(prow + 8 * g)          = w0a;
  *(v4f*)(prow + 8 * g + 4)      = w0b;
  *(v4f*)(prow + 16 + 8 * g)     = w1a;
  *(v4f*)(prow + 16 + 8 * g + 4) = w1b;
  if (g == 0) {
    const size_t mi = (size_t)sp * (BATCH * N_HEADS * NQ) +
                      (size_t)(b * N_HEADS + h) * NQ + q0 + lr;
    Pm[mi] = mrun;
    Pl[mi] = lrun;
  }
}

// ---------------------------------------------------------------------------
// Combine kv-split partials: O = sum_s O_s * e^{m_s - m} / sum_s l_s * e^{m_s - m}
// One thread per (b,h,q,d); writes f16 [B][NQ][256] for the output projection.
// ---------------------------------------------------------------------------
__global__ void combine_kernel(const float* __restrict__ Po, const float* __restrict__ Pm,
                               const float* __restrict__ Pl, half_t* __restrict__ Oh) {
  const int idx = blockIdx.x * blockDim.x + threadIdx.x;
  const int total = BATCH * N_HEADS * NQ * D_HEAD;
  if (idx >= total) return;
  const int qidx = idx >> 5;
  const int d = idx & 31;
  const int OS = BATCH * N_HEADS * NQ * D_HEAD;
  const int MS = BATCH * N_HEADS * NQ;
  float m = -1e30f;
#pragma unroll
  for (int s = 0; s < KV_SPLIT; ++s) m = fmaxf(m, Pm[s * MS + qidx]);
  float l = 0.f, o = 0.f;
#pragma unroll
  for (int s = 0; s < KV_SPLIT; ++s) {
    const float w = __expf(Pm[s * MS + qidx] - m);
    l = fmaf(Pl[s * MS + qidx], w, l);
    o = fmaf(Po[(size_t)s * OS + idx], w, o);
  }
  const int q = qidx % NQ, bh = qidx / NQ, h = bh % N_HEADS, b = bh / N_HEADS;
  Oh[(size_t)(b * NQ + q) * D_MODEL + h * D_HEAD + d] = (half_t)(o / l);
}

// ---------------------------------------------------------------------------
extern "C" void kernel_launch(void* const* d_in, const int* in_sizes, int n_in,
                              void* d_out, int out_size, void* d_ws, size_t ws_size,
                              hipStream_t stream) {
  const float* q    = (const float*)d_in[0];
  const float* kv   = (const float*)d_in[1];
  const float* bias = (const float*)d_in[2];
  const float* Wq   = (const float*)d_in[3];
  const float* Wk   = (const float*)d_in[4];
  const float* Wv   = (const float*)d_in[5];
  const float* Wo   = (const float*)d_in[6];
  const float* gq   = (const float*)d_in[7];
  const float* bq   = (const float*)d_in[8];
  const float* gkv  = (const float*)d_in[9];
  const float* bkv  = (const float*)d_in[10];
  float* out = (float*)d_out;

  char* ws = (char*)d_ws;
  size_t off = 0;
  auto alloc = [&](size_t bytes) {
    void* p = ws + off;
    off = (off + bytes + 255) & ~(size_t)255;
    return p;
  };
  const int WN = D_MODEL * D_MODEL;  // 65536
  half_t* wqh = (half_t*)alloc(WN * 2);
  half_t* wkh = (half_t*)alloc(WN * 2);
  half_t* wvh = (half_t*)alloc(WN * 2);
  half_t* woh = (half_t*)alloc(WN * 2);
  half_t* qn  = (half_t*)alloc((size_t)BATCH * NQ  * D_MODEL * 2);
  half_t* kvn = (half_t*)alloc((size_t)BATCH * NKV * D_MODEL * 2);
  half_t* Qp  = (half_t*)alloc((size_t)BATCH * NQ  * D_MODEL * 2);
  half_t* Kp  = (half_t*)alloc((size_t)BATCH * NKV * D_MODEL * 2);
  half_t* Vt  = (half_t*)alloc((size_t)BATCH * NKV * D_MODEL * 2);
  half_t* Ao  = (half_t*)alloc((size_t)BATCH * NQ  * D_MODEL * 2);
  float*  Po  = (float*)alloc((size_t)KV_SPLIT * BATCH * N_HEADS * NQ * D_HEAD * 4);
  float*  Pm  = (float*)alloc((size_t)KV_SPLIT * BATCH * N_HEADS * NQ * 4);
  float*  Pl  = (float*)alloc((size_t)KV_SPLIT * BATCH * N_HEADS * NQ * 4);

  // 1. Weights to f16.
  cvt_kernel<<<WN / 256, 256, 0, stream>>>(Wq, wqh, WN);
  cvt_kernel<<<WN / 256, 256, 0, stream>>>(Wk, wkh, WN);
  cvt_kernel<<<WN / 256, 256, 0, stream>>>(Wv, wvh, WN);
  cvt_kernel<<<WN / 256, 256, 0, stream>>>(Wo, woh, WN);

  // 2. LayerNorm (one wave per row).
  ln_kernel<<<(BATCH * NQ  * 32) / 256, 256, 0, stream>>>(q,  gq,  bq,  qn,  BATCH * NQ);
  ln_kernel<<<(BATCH * NKV * 32) / 256, 256, 0, stream>>>(kv, gkv, bkv, kvn, BATCH * NKV);

  // 3. Projections (WMMA GEMMs). V is stored transposed per head.
  gemm_kernel<0><<<BATCH * NQ  / 16, 512, 0, stream>>>(qn,  wqh, Qp, nullptr, nullptr);
  gemm_kernel<0><<<BATCH * NKV / 16, 512, 0, stream>>>(kvn, wkh, Kp, nullptr, nullptr);
  gemm_kernel<1><<<BATCH * NKV / 16, 512, 0, stream>>>(kvn, wvh, Vt, nullptr, nullptr);

  // 4. Flash attention, kv split 4 ways for occupancy (4096 waves); bias
  //    (537 MB) streamed exactly once; K/V re-reads are L2-resident (16 MB).
  dim3 ag((NQ / 16 / 8) * KV_SPLIT, N_HEADS, BATCH);
  attn_kernel<<<ag, 256, 0, stream>>>(Qp, Kp, Vt, bias, Po, Pm, Pl);

  // 5. Merge kv-split partials.
  const int total = BATCH * N_HEADS * NQ * D_HEAD;
  combine_kernel<<<(total + 255) / 256, 256, 0, stream>>>(Po, Pm, Pl, Ao);

  // 6. Output projection + residual, f32 out.
  gemm_kernel<2><<<BATCH * NQ / 16, 512, 0, stream>>>(Ao, woh, nullptr, q, out);
}
